// GNBlock_36593121362173
// MI455X (gfx1250) — matrix operand order repaired and checked by reading, not compiled, not run on hardware
//
#include <hip/hip_runtime.h>

// ---------------------------------------------------------------------------
// Types for WMMA bf16 (CDNA5 / gfx1250, wave32)
// ---------------------------------------------------------------------------
typedef __attribute__((ext_vector_type(16))) __bf16         v16bf;
typedef __attribute__((ext_vector_type(8)))  float          v8f;
typedef __attribute__((ext_vector_type(4)))  unsigned int   u32x4;
typedef __attribute__((ext_vector_type(4)))  unsigned short u16x4;

union FragU { v16bf v; u32x4 q[2]; };

__device__ __forceinline__ unsigned short f2bf(float f) {
  unsigned int u = __float_as_uint(f);
  u += 0x7FFFu + ((u >> 16) & 1u);   // round-to-nearest-even
  return (unsigned short)(u >> 16);
}

__device__ __forceinline__ u16x4 f2bf4(float4 f) {
  u16x4 r;
  r[0] = f2bf(f.x); r[1] = f2bf(f.y); r[2] = f2bf(f.z); r[3] = f2bf(f.w);
  return r;
}

// ---------------------------------------------------------------------------
// One MLP layer: D[RT*16, NOUT] = relu?(X[RT*16, K] * W[K,NOUT] + b)
//  - X bf16 in LDS (row-major, stride ldIn); Wt bf16 global TRANSPOSED
//    (Wt[n*K+k]) so B fragments are contiguous-K loads per lane.
//  - Register-blocked: each wave owns one n-tile and ALL RT row-tiles
//    (RT accumulators), so each B fragment feeds RT WMMAs (4x less global
//    weight traffic, RT independent accumulation chains per k-step).
//  - A fragment: lane holds row lane%16; half-wave 0 K {0..15},
//    half-wave 1 K {16..31}. B: lane holds col lane%16, K-contig per
//    half-wave. C/D: col = lane%16, row = v + 8*(lane/16).
// ---------------------------------------------------------------------------
template<int K, int NOUT, int RT, bool RELU, bool LAST>
__device__ __forceinline__ void gemm_layer(
    const unsigned short* Xin, int ldIn,
    const unsigned short* __restrict__ Wt, const float* __restrict__ bias,
    unsigned short* Xout, int ldOut,
    float* __restrict__ Gout, long long rowBase, long long rowMax)
{
  const int w    = threadIdx.x >> 5;
  const int lane = threadIdx.x & 31;
  const int lh   = lane >> 4;
  const int lm   = lane & 15;
  for (int nt = w; nt < NOUT / 16; nt += 8) {
    const int n0 = nt * 16;
    const float bv = bias[n0 + lm];
    v8f c[RT];
#pragma unroll
    for (int r = 0; r < RT; ++r) c[r] = v8f{ bv, bv, bv, bv, bv, bv, bv, bv };
    const unsigned short* brow = Wt + (size_t)(n0 + lm) * K + lh * 16;
    const unsigned short* arow = Xin + lm * ldIn + lh * 16;
#pragma unroll 4
    for (int k0 = 0; k0 < K; k0 += 32) {
      FragU fb;
      const u32x4* pb = (const u32x4*)(brow + k0);
      fb.q[0] = pb[0];      // K = k0+16*lh    .. +7
      fb.q[1] = pb[1];      // K = k0+16*lh+8  .. +15
#pragma unroll
      for (int r = 0; r < RT; ++r) {
        FragU fa;
        const u32x4* pa = (const u32x4*)(arow + r * 16 * ldIn + k0);
        fa.q[0] = pa[0];    // K = k0+16*lh    .. +7
        fa.q[1] = pa[1];    // K = k0+16*lh+8  .. +15
        c[r] = __builtin_amdgcn_wmma_f32_16x16x32_bf16(
                   false, fa.v, false, fb.v, (short)0, c[r], false, false);
      }
    }
#pragma unroll
    for (int r = 0; r < RT; ++r) {
#pragma unroll
      for (int v = 0; v < 8; ++v) {
        float val = c[r][v];
        if (RELU) val = val > 0.0f ? val : 0.0f;
        const int row = r * 16 + lh * 8 + v;
        const int col = n0 + lm;
        if (!LAST) {
          Xout[row * ldOut + col] = f2bf(val);
        } else {
          const long long gr = rowBase + row;
          if (gr < rowMax) Gout[gr * (long long)NOUT + col] = val;
        }
      }
    }
  }
}

// ---------------------------------------------------------------------------
// Utility kernels
// ---------------------------------------------------------------------------
__global__ void k_zero(float* __restrict__ p, long long n) {
  long long i = (long long)blockIdx.x * 256 + threadIdx.x;
  if (i < n) p[i] = 0.0f;
}

// fp32 W[K,N] row-major -> bf16 Wt[N,K] row-major (4 k's per thread, 8B store)
__global__ void k_convert(const float* __restrict__ W,
                          unsigned short* __restrict__ Wt, int K, int N) {
  int i = blockIdx.x * 256 + threadIdx.x;
  if (i >= (K * N) >> 2) return;
  const int idx = i * 4;
  const int n = idx / K, k0 = idx - n * K;
  u16x4 r;
#pragma unroll
  for (int j = 0; j < 4; ++j) r[j] = f2bf(W[(size_t)(k0 + j) * N + n]);
  *(u16x4*)(Wt + idx) = r;
}

// ---------------------------------------------------------------------------
// Edge MLP: 64 edges/block.  x = [g | feats[src] | feats[dst] | efeats] (512)
// 512 -> 256 -> 256 -> 256 -> 128, writes fp32 'a' into d_out hn slot.
// ---------------------------------------------------------------------------
__global__ __launch_bounds__(256) void k_edge_mlp(
    const float* __restrict__ feats, const float* __restrict__ efeats,
    const float* __restrict__ gfeat, const int* __restrict__ src,
    const int* __restrict__ dst, const unsigned short* __restrict__ We,
    const float* __restrict__ b0, const float* __restrict__ b1,
    const float* __restrict__ b2, const float* __restrict__ b3,
    float* __restrict__ aout, int E)
{
  extern __shared__ unsigned short sh[];
  unsigned short* bufA = sh;              // 64 x 512 bf16
  unsigned short* bufB = sh + 64 * 512;   // 64 x 256 bf16
  const long long e0 = (long long)blockIdx.x * 64;
  for (int i = threadIdx.x; i < (64 * 512) / 4; i += 256) {
    const int idx = i * 4;
    const int r = idx >> 9, k = idx & 511;
    long long e = e0 + r; if (e >= E) e = E - 1;
    float4 v;
    if (k < 128)      v = *(const float4*)(gfeat + k);
    else if (k < 256) v = *(const float4*)(feats + (size_t)src[e] * 128 + (k - 128));
    else if (k < 384) v = *(const float4*)(feats + (size_t)dst[e] * 128 + (k - 256));
    else              v = *(const float4*)(efeats + (size_t)e * 128 + (k - 384));
    *(u16x4*)(bufA + idx) = f2bf4(v);
  }
  __syncthreads();
  gemm_layer<512,256,4,true ,false>(bufA,512, We,          b0, bufB,256, nullptr,0,0);
  __syncthreads();
  gemm_layer<256,256,4,true ,false>(bufB,256, We+131072,   b1, bufA,512, nullptr,0,0);
  __syncthreads();
  gemm_layer<256,256,4,true ,false>(bufA,512, We+196608,   b2, bufB,256, nullptr,0,0);
  __syncthreads();
  gemm_layer<256,128,4,false,true >(bufB,256, We+262144,   b3, nullptr,0, aout, e0, E);
}

// ---------------------------------------------------------------------------
// Per-type sums/counts of 'a' (LDS-atomic partials -> global atomics) and
// per-(dst-node, type) edge counts (one int atomic per edge).
// ---------------------------------------------------------------------------
__global__ __launch_bounds__(256) void k_type_stats(
    const float* __restrict__ a, const int* __restrict__ ei,
    const int* __restrict__ dst, float* __restrict__ type_sum,
    float* __restrict__ counts, int* __restrict__ cnt, int E, int epb)
{
  __shared__ float sts[8 * 128];
  __shared__ float scn[8];
  for (int i = threadIdx.x; i < 1024; i += 256) sts[i] = 0.0f;
  if (threadIdx.x < 8) scn[threadIdx.x] = 0.0f;
  __syncthreads();
  const int c    = threadIdx.x & 127;
  const int half = threadIdx.x >> 7;
  long long ebeg = (long long)blockIdx.x * epb;
  long long eend = ebeg + epb; if (eend > E) eend = E;
  for (long long e = ebeg + half; e < eend; e += 2) {
    const int t = ei[e];
    atomicAdd(&sts[t * 128 + c], a[e * 128 + c]);
    if (c == 0) {
      atomicAdd(&scn[t], 1.0f);
      atomicAdd(&cnt[(size_t)dst[e] * 8 + t], 1);
    }
  }
  __syncthreads();
  for (int i = threadIdx.x; i < 1024; i += 256) atomicAdd(&type_sum[i], sts[i]);
  if (threadIdx.x < 8) atomicAdd(&counts[threadIdx.x], scn[threadIdx.x]);
}

__global__ void k_tfeats(const float* __restrict__ type_sum,
                         const float* __restrict__ counts,
                         float* __restrict__ t_feats) {
  int i = blockIdx.x * 256 + threadIdx.x;
  if (i < 1024) t_feats[i] = type_sum[i] / counts[i >> 7];
}

// ef[n][c] = sum_t cnt[n][t] * t_feats[t][c]   (dense, replaces 41M atomics)
__global__ void k_ef(const int* __restrict__ cnt,
                     const float* __restrict__ t_feats,
                     float* __restrict__ ef, int N) {
  long long i = (long long)blockIdx.x * 256 + threadIdx.x;
  if (i >= (long long)N * 128) return;
  const int n = (int)(i >> 7), c = (int)(i & 127);
  float s = 0.0f;
#pragma unroll
  for (int t = 0; t < 8; ++t) s += (float)cnt[n * 8 + t] * t_feats[t * 128 + c];
  ef[i] = s;
}

// hn[e] = t_feats[ei[e]]  (float4 copies; overwrites 'a' after consumption)
__global__ void k_hn(const int* __restrict__ ei,
                     const float* __restrict__ t_feats,
                     float4* __restrict__ hn, long long total4) {
  long long i = (long long)blockIdx.x * 256 + threadIdx.x;
  if (i >= total4) return;
  const long long e = i >> 5; const int c4 = (int)(i & 31);
  hn[i] = *(const float4*)(t_feats + (size_t)ei[e] * 128 + c4 * 4);
}

// ---------------------------------------------------------------------------
// Node MLP: 64 nodes/block.  x = [g | feats[n] | ef[n]] (384)
// ---------------------------------------------------------------------------
__global__ __launch_bounds__(256) void k_node_mlp(
    const float* __restrict__ feats, const float* __restrict__ gfeat,
    const float* __restrict__ ef, const unsigned short* __restrict__ Wn,
    const float* __restrict__ b0, const float* __restrict__ b1,
    const float* __restrict__ b2, const float* __restrict__ b3,
    float* __restrict__ nnout, int N)
{
  extern __shared__ unsigned short sh[];
  unsigned short* bufA = sh;              // 64 x 384 bf16
  unsigned short* bufB = sh + 64 * 384;   // 64 x 256 bf16
  const long long n0 = (long long)blockIdx.x * 64;
  for (int i = threadIdx.x; i < (64 * 384) / 4; i += 256) {
    const int idx = i * 4;
    const int r = idx / 384, k = idx - r * 384;
    long long n = n0 + r; if (n >= N) n = N - 1;
    float4 v;
    if (k < 128)      v = *(const float4*)(gfeat + k);
    else if (k < 256) v = *(const float4*)(feats + (size_t)n * 128 + (k - 128));
    else              v = *(const float4*)(ef + (size_t)n * 128 + (k - 256));
    *(u16x4*)(bufA + idx) = f2bf4(v);
  }
  __syncthreads();
  gemm_layer<384,256,4,true ,false>(bufA,384, Wn,          b0, bufB,256, nullptr,0,0);
  __syncthreads();
  gemm_layer<256,256,4,true ,false>(bufB,256, Wn+ 98304,   b1, bufA,384, nullptr,0,0);
  __syncthreads();
  gemm_layer<256,256,4,true ,false>(bufA,384, Wn+163840,   b2, bufB,256, nullptr,0,0);
  __syncthreads();
  gemm_layer<256,128,4,false,true >(bufB,256, Wn+229376,   b3, nullptr,0, nnout, n0, N);
}

// nsum[c] = sum_n nn[n][c]; one block per column, tree reduce (no atomics)
__global__ __launch_bounds__(256) void k_nsum(const float* __restrict__ nn,
                                              float* __restrict__ nsum, int N) {
  __shared__ float s[256];
  const int c = blockIdx.x;
  float acc = 0.0f;
  for (int n = threadIdx.x; n < N; n += 256) acc += nn[(size_t)n * 128 + c];
  s[threadIdx.x] = acc; __syncthreads();
  for (int off = 128; off > 0; off >>= 1) {
    if (threadIdx.x < off) s[threadIdx.x] += s[threadIdx.x + off];
    __syncthreads();
  }
  if (threadIdx.x == 0) nsum[c] = s[0];
}

// ---------------------------------------------------------------------------
// Graph MLP: one workgroup, one valid row. x = [g | nsum | sum_t type_sum[t]]
// (hn.sum(0) == column-sum of type_sum, since hn = gathered t_feats)
// ---------------------------------------------------------------------------
__global__ __launch_bounds__(256) void k_graph_mlp(
    const float* __restrict__ gfeat, const float* __restrict__ nsum,
    const float* __restrict__ type_sum, const unsigned short* __restrict__ Wg,
    const float* __restrict__ b0, const float* __restrict__ b1,
    const float* __restrict__ b2, const float* __restrict__ b3,
    float* __restrict__ gn)
{
  __shared__ unsigned short bufA[16 * 384];
  __shared__ unsigned short bufB[16 * 256];
  for (int i = threadIdx.x; i < 16 * 384; i += 256) {
    const int r = i / 384, k = i - r * 384;
    float v = 0.0f;
    if (r == 0) {
      if (k < 128)      v = gfeat[k];
      else if (k < 256) v = nsum[k - 128];
      else {
        float s = 0.0f;
#pragma unroll
        for (int t = 0; t < 8; ++t) s += type_sum[t * 128 + (k - 256)];
        v = s;
      }
    }
    bufA[i] = f2bf(v);
  }
  __syncthreads();
  gemm_layer<384,256,1,true ,false>(bufA,384, Wg,          b0, bufB,256, nullptr,0,0);
  __syncthreads();
  gemm_layer<256,256,1,true ,false>(bufB,256, Wg+ 98304,   b1, bufA,384, nullptr,0,0);
  __syncthreads();
  gemm_layer<256,256,1,true ,false>(bufA,384, Wg+163840,   b2, bufB,256, nullptr,0,0);
  __syncthreads();
  gemm_layer<256,128,1,false,true >(bufB,256, Wg+229376,   b3, nullptr,0, gn, 0, 1);
}

// ---------------------------------------------------------------------------
// Host launch
// ---------------------------------------------------------------------------
extern "C" void kernel_launch(void* const* d_in, const int* in_sizes, int n_in,
                              void* d_out, int out_size, void* d_ws, size_t ws_size,
                              hipStream_t stream)
{
  const float* feats  = (const float*)d_in[0];
  const float* efeats = (const float*)d_in[1];
  const float* gfeat  = (const float*)d_in[2];
  const int*   src    = (const int*)d_in[3];
  const int*   dst    = (const int*)d_in[4];
  const int*   ei     = (const int*)d_in[5];
  const int N = in_sizes[0] / 128;
  const int E = in_sizes[1] / 128;

  const float* We32[4] = {(const float*)d_in[ 6], (const float*)d_in[ 8],
                          (const float*)d_in[10], (const float*)d_in[12]};
  const float* be[4]   = {(const float*)d_in[ 7], (const float*)d_in[ 9],
                          (const float*)d_in[11], (const float*)d_in[13]};
  const float* Wn32[4] = {(const float*)d_in[14], (const float*)d_in[16],
                          (const float*)d_in[18], (const float*)d_in[20]};
  const float* bn[4]   = {(const float*)d_in[15], (const float*)d_in[17],
                          (const float*)d_in[19], (const float*)d_in[21]};
  const float* Wg32[4] = {(const float*)d_in[22], (const float*)d_in[24],
                          (const float*)d_in[26], (const float*)d_in[28]};
  const float* bg[4]   = {(const float*)d_in[23], (const float*)d_in[25],
                          (const float*)d_in[27], (const float*)d_in[29]};

  // workspace layout
  unsigned short* We = (unsigned short*)d_ws;   // 294912 bf16
  unsigned short* Wn = We + 294912;             // 262144 bf16
  unsigned short* Wg = Wn + 262144;             // 262144 bf16
  float* F        = (float*)(Wg + 262144);
  float* type_sum = F;                          // 1024
  float* counts   = F + 1024;                   // 8 (padded to 32)
  float* t_feats  = F + 1056;                   // 1024
  float* nsum     = F + 2080;                   // 128
  int*   cnt      = (int*)(F + 2208);           // N*8
  float* ef       = F + 2208 + (size_t)N * 8;   // N*128

  float* nn = (float*)d_out;
  float* hn = nn + (size_t)N * 128;
  float* gn = hn + (size_t)E * 128;

  // 1) zero the accumulators (every call; harness doesn't re-poison)
  k_zero<<<(1056 + 255) / 256, 256, 0, stream>>>(type_sum, 1056);
  k_zero<<<(unsigned)(((long long)N * 8 + 255) / 256), 256, 0, stream>>>((float*)cnt, (long long)N * 8);

  // 2) fp32 -> bf16 transposed weights
  {
    const int eK[4] = {512,256,256,256}, eN[4] = {256,256,256,128};
    const int nK[4] = {384,256,256,256}, nN[4] = {256,256,256,128};
    size_t off = 0;
    for (int l = 0; l < 4; ++l) { int tot = eK[l]*eN[l];
      k_convert<<<(tot/4+255)/256, 256, 0, stream>>>(We32[l], We+off, eK[l], eN[l]); off += tot; }
    off = 0;
    for (int l = 0; l < 4; ++l) { int tot = nK[l]*nN[l];
      k_convert<<<(tot/4+255)/256, 256, 0, stream>>>(Wn32[l], Wn+off, nK[l], nN[l]); off += tot; }
    off = 0;
    for (int l = 0; l < 4; ++l) { int tot = nK[l]*nN[l];
      k_convert<<<(tot/4+255)/256, 256, 0, stream>>>(Wg32[l], Wg+off, nK[l], nN[l]); off += tot; }
  }

  // 3) edge MLP -> 'a' into hn slot (fp32)
  {
    size_t shE = (size_t)(64*512 + 64*256) * sizeof(unsigned short); // 96 KB (<320KB/WGP)
    k_edge_mlp<<<(unsigned)((E + 63) / 64), 256, shE, stream>>>(
        feats, efeats, gfeat, src, dst, We, be[0], be[1], be[2], be[3], hn, E);
  }

  // 4) per-type sums/counts + per-(dst,type) counts
  {
    const int epb = 512;
    k_type_stats<<<(unsigned)((E + epb - 1) / epb), 256, 0, stream>>>(
        hn, ei, dst, type_sum, counts, cnt, E, epb);
  }

  // 5) t_feats = type_sum / counts
  k_tfeats<<<4, 256, 0, stream>>>(type_sum, counts, t_feats);

  // 6) ef (dense cnt x t_feats contraction)
  k_ef<<<(unsigned)(((long long)N * 128 + 255) / 256), 256, 0, stream>>>(cnt, t_feats, ef, N);

  // 7) hn = gather(t_feats, ei) — overwrites 'a'
  {
    long long tot4 = (long long)E * 32;
    k_hn<<<(unsigned)((tot4 + 255) / 256), 256, 0, stream>>>(ei, t_feats, (float4*)hn, tot4);
  }

  // 8) node MLP -> nn
  {
    size_t shN = (size_t)(64*384 + 64*256) * sizeof(unsigned short); // 80 KB
    k_node_mlp<<<(unsigned)((N + 63) / 64), 256, shN, stream>>>(
        feats, gfeat, ef, Wn, bn[0], bn[1], bn[2], bn[3], nn, N);
  }

  // 9) nn column sums
  k_nsum<<<128, 256, 0, stream>>>(nn, nsum, N);

  // 10) graph MLP -> gn
  k_graph_mlp<<<1, 256, 0, stream>>>(gfeat, nsum, type_sum, Wg,
                                     bg[0], bg[1], bg[2], bg[3], gn);

  (void)n_in; (void)out_size; (void)ws_size;
}
